// LCM_38019050505053
// MI455X (gfx1250) — matrix operand order repaired and checked by reading, not compiled, not run on hardware
//
#include <hip/hip_runtime.h>
#include <hip/hip_bf16.h>

typedef __attribute__((ext_vector_type(16))) _Float16 v16h;
typedef __attribute__((ext_vector_type(8)))  _Float16 v8h;
typedef __attribute__((ext_vector_type(8)))  float    v8f;

#define NB 16384      // batch
#define ND 16         // datasets
#define KP 192        // padded K (input dim 180 -> 192)
#define LSTRIDE 200   // LDS K-stride (halfs) for M^T columns (bank-conflict padding)

// ---------------- kernel 0: zero bucket counters ----------------
__global__ void lcm_zero(int* counts) {
    if (threadIdx.x < ND) counts[threadIdx.x] = 0;
}

// ---------------- kernel 1: build M^T (f16) per dataset ----------------
// Mtg layout: [d][j(180)][i stride 192] f16 (transposed M), so the GEMM
// B-operand (K-major per output column) is contiguous.
__global__ void lcm_build_m(const float* __restrict__ us,
                            const float* __restrict__ vs,
                            const float* __restrict__ zs,
                            _Float16* __restrict__ Mtg) {
    int gid = blockIdx.x * blockDim.x + threadIdx.x;   // 15 blocks * 192 = 2880
    int d = gid / 180, i = gid % 180;
    int g = i / 90;
    float a = (float)(i % 90 + 1);
    const float* u = us + d * 6 + g * 3;
    const float* v = vs + d * 4 + g * 2;
    const float* z = zs + d * 6 + g * 3;
    float mu = u[0] + a * u[1] + a * a * u[2];
    float sg = v[0] + a * v[1];
    float gm = z[0] + a * z[1] + a * a * z[2];
    float pg1 = 1.0f / (1.0f + __expf(-gm));   // sigmoid(+gamma)
    float pg0 = 1.0f - pg1;                    // sigmoid(-gamma)
    float q = -0.5f / (sg * sg + 1e-8f);
    // logits are q*(mu-a')^2, q<0 -> max at a' nearest mu (clamped to [1,90])
    float as = fminf(fmaxf(rintf(mu), 1.0f), 90.0f);
    float lmax = q * (mu - as) * (mu - as);
    float sum = 0.0f;
    for (int ap = 1; ap <= 90; ++ap) {
        float dd = mu - (float)ap;
        sum += __expf(q * dd * dd - lmax);
    }
    float isum = 1.0f / sum;
    _Float16* col = Mtg + (size_t)d * (180 * KP);
    for (int ap = 1; ap <= 90; ++ap) {
        float dd = mu - (float)ap;
        float p = __expf(q * dd * dd - lmax) * isum;
        col[(size_t)(ap - 1) * KP + i]      = (_Float16)(p * pg0);
        col[(size_t)(90 + ap - 1) * KP + i] = (_Float16)(p * pg1);
    }
}

// ---------------- kernel 2: row softmax -> f16 P[B][192] (zero padded) -------
__global__ __launch_bounds__(256) void lcm_softmax(const float* __restrict__ x,
                                                   _Float16* __restrict__ Ph) {
    int lane = threadIdx.x & 31;
    int b = (int)((blockIdx.x * blockDim.x + threadIdx.x) >> 5); // one wave per row
    const float* xr = x + (size_t)b * 180;
    float v[6], mx = -3.4e38f;
#pragma unroll
    for (int j = 0; j < 6; ++j) {
        int k = lane + 32 * j;
        v[j] = (k < 180) ? xr[k] : -3.4e38f;
        mx = fmaxf(mx, v[j]);
    }
#pragma unroll
    for (int m = 1; m < 32; m <<= 1) mx = fmaxf(mx, __shfl_xor(mx, m));
    float e[6], s = 0.0f;
#pragma unroll
    for (int j = 0; j < 6; ++j) {
        int k = lane + 32 * j;
        e[j] = (k < 180) ? __expf(v[j] - mx) : 0.0f;
        s += e[j];
    }
#pragma unroll
    for (int m = 1; m < 32; m <<= 1) s += __shfl_xor(s, m);
    float inv = 1.0f / s;
    _Float16* pr = Ph + (size_t)b * KP;
#pragma unroll
    for (int j = 0; j < 6; ++j) pr[lane + 32 * j] = (_Float16)(e[j] * inv);
}

// ---------------- kernel 3: bucket rows by dataset ----------------
__global__ void lcm_bucket(const int* __restrict__ dnum,
                           int* __restrict__ counts,
                           int* __restrict__ buckets) {
    int b = blockIdx.x * blockDim.x + threadIdx.x;
    if (b >= NB) return;
    int d = dnum[b] & (ND - 1);
    int pos = atomicAdd(&counts[d], 1);
    buckets[d * NB + pos] = b;
}

// ---------------- kernel 4: WMMA GEMM + normalize ----------------
// grid = (256 chunks of 64 rows, 16 datasets), block = 128 (4 waves).
// Wave w: 16-row A tile, 12 N-tiles of 16 -> 72 v_wmma_f32_16x16x32_f16,
// with all A fragments preloaded and B fragments software-pipelined depth-2.
__global__ __launch_bounds__(128) void lcm_gemm(const _Float16* __restrict__ P,
                                                const _Float16* __restrict__ Mtg,
                                                const int* __restrict__ counts,
                                                const int* __restrict__ buckets,
                                                float* __restrict__ out) {
    __shared__ _Float16 MtL[KP * LSTRIDE];  // 192 cols * 200 halfs = 75 KB
    __shared__ int rowsL[4][16];

    const int d = blockIdx.y;
    const int count = counts[d];
    const int base = blockIdx.x * 64;
    if (base >= count) return;             // block-uniform: EXEC stays all-ones

    const int tid = threadIdx.x;
    const int wave = tid >> 5;
    const int lane = tid & 31;
    const int lm = lane & 15;
    const int half = lane >> 4;

    // row ids for this wave's 16-row tile (issued first: feeds A preloads)
    int idx = base + wave * 16 + lm;
    int cl = (idx < count) ? idx : (count - 1);
    int row = buckets[d * NB + cl];
    if (lane < 16) rowsL[wave][lm] = (idx < count) ? row : -1;

    // Preload all 6 A fragments (overlaps with LDS staging below).
    // A fragment, contiguous K order as in B: lanes 0-15 K={0..15}, lanes 16-31 K={16..31}
    const _Float16* prow = P + (size_t)row * KP;
    v8h a0[6], a1[6];
#pragma unroll
    for (int s = 0; s < 6; ++s) {
        const v8h* pa = (const v8h*)(prow + s * 32 + half * 16);
        a0[s] = pa[0];
        a1[s] = pa[1];   // +8 halfs
    }

    // Stage M^T[d] into LDS (64-bit chunks); pad n>=180 and K>=180 with zeros.
    {
        typedef __attribute__((ext_vector_type(2))) unsigned u2;
        const u2* gsrc = (const u2*)(Mtg + (size_t)d * (180 * KP));
        u2* ldst = (u2*)MtL;
        const u2 zz = {0u, 0u};
#pragma unroll 2
        for (int e = tid; e < KP * (KP / 4); e += 128) {     // 192 cols * 48 qwords
            int n = e / (KP / 4), kq = e % (KP / 4);         // kq = 4-half chunk in K
            u2 val = (n < 180 && kq < 45) ? gsrc[n * (KP / 4) + kq] : zz;
            ldst[n * (LSTRIDE / 4) + kq] = val;              // LSTRIDE/4 = 50 qwords
        }
    }
    __syncthreads();

    v8f acc[12] = {};

    // B fragment loader: lane holds column n = t*16+lm, K = half*16 + 0..15
    auto loadB = [&](int it) -> v16h {
        int s = it / 12, t = it % 12;
        const v8h* pb = (const v8h*)(&MtL[(t * 16 + lm) * LSTRIDE + s * 32 + half * 16]);
        return __builtin_shufflevector(pb[0], pb[1], 0, 1, 2, 3, 4, 5, 6, 7,
                                       8, 9, 10, 11, 12, 13, 14, 15);
    };

    // Depth-2 software pipeline over the flattened 6x12 (s,t) iteration space.
    v16h b_p0 = loadB(0);
    v16h b_p1 = loadB(1);
#pragma unroll
    for (int it = 0; it < 72; ++it) {
        int s = it / 12, t = it % 12;
        v16h Bc = b_p0;
        b_p0 = b_p1;
        if (it + 2 < 72) b_p1 = loadB(it + 2);
        v16h A = __builtin_shufflevector(a0[s], a1[s], 0, 1, 2, 3, 4, 5, 6, 7,
                                         8, 9, 10, 11, 12, 13, 14, 15);
        acc[t] = __builtin_amdgcn_wmma_f32_16x16x32_f16(
            false, A, false, Bc, (short)0, acc[t], false, false);
    }

    // Row sums (padded columns are exactly 0); reduce within 16-lane halves
    // (xor masks < 16 never cross the M / M+8 halves of the C layout).
    float inv[8];
#pragma unroll
    for (int r = 0; r < 8; ++r) {
        float ps = 0.0f;
#pragma unroll
        for (int t = 0; t < 12; ++t) ps += acc[t][r];
        ps += __shfl_xor(ps, 1);
        ps += __shfl_xor(ps, 2);
        ps += __shfl_xor(ps, 4);
        ps += __shfl_xor(ps, 8);
        inv[r] = 1.0f / ps;
    }
#pragma unroll
    for (int t = 0; t < 12; ++t) {
        int j = t * 16 + lm;
        if (j < 180) {
#pragma unroll
            for (int r = 0; r < 8; ++r) {
                int ri = rowsL[wave][r + half * 8];   // C layout: row = r + half*8
                if (ri >= 0) out[(size_t)ri * 180 + j] = acc[t][r] * inv[r];
            }
        }
    }
}

extern "C" void kernel_launch(void* const* d_in, const int* in_sizes, int n_in,
                              void* d_out, int out_size, void* d_ws, size_t ws_size,
                              hipStream_t stream) {
    const float* x    = (const float*)d_in[0];
    const int*   dnum = (const int*)d_in[1];
    const float* us   = (const float*)d_in[2];
    const float* vs   = (const float*)d_in[3];
    const float* zs   = (const float*)d_in[4];
    float* out = (float*)d_out;

    char* ws = (char*)d_ws;
    int* counts       = (int*)ws;                                    // 64 B
    int* buckets      = (int*)(ws + 256);                            // 1 MB
    _Float16* Mtg     = (_Float16*)(ws + 256 + (size_t)ND * NB * 4); // 1.05 MB
    _Float16* Ph      = (_Float16*)(ws + 256 + (size_t)ND * NB * 4
                                    + (size_t)ND * 180 * KP * 2);    // 6 MB

    lcm_zero<<<1, 32, 0, stream>>>(counts);
    lcm_build_m<<<15, 192, 0, stream>>>(us, vs, zs, Mtg);
    lcm_softmax<<<NB / 8, 256, 0, stream>>>(x, Ph);
    lcm_bucket<<<NB / 256, 256, 0, stream>>>(dnum, counts, buckets);
    lcm_gemm<<<dim3(256, 16), 128, 0, stream>>>(Ph, Mtg, counts, buckets, out);
}
